// CUP_28054726378228
// MI455X (gfx1250) — matrix operand order repaired and checked by reading, not compiled, not run on hardware
//
#include <hip/hip_runtime.h>
#include <hip/hip_bf16.h>

typedef __attribute__((ext_vector_type(16))) __bf16 v16bf;
typedef __attribute__((ext_vector_type(8)))  float  v8f;

#define BSZ   4
#define CCH   64
#define OCH   64
#define LVL   4
#define IH    128
#define IW    128
#define KDIM  576          // C * 9
#define NTAPS 288          // 32 pixels * 9 taps
#define KFUSE 256          // L * O for the fused 1x1
#define NKT_D 18           // K-tiles for deform/offset conv (576/32)
#define NKT_F 8            // K-tiles for fuse (256/32)

// ---------- bf16 helpers (RNE) ----------
static __device__ __forceinline__ unsigned short f2bf_bits(float f) {
  union { float f; unsigned u; } a; a.f = f;
  unsigned r = a.u + 0x7FFFu + ((a.u >> 16) & 1u);
  return (unsigned short)(r >> 16);
}

// Involution on K index: swap bits 3 and 4. Storing A at slot s = kswap(K)
// makes each lane's 16 WMMA A-elements contiguous in LDS (2x ds_load_b128),
// given the ISA 16-bit A striping (lane-half hi holds K {e+8hi} u {e+8hi+16}).
static __device__ __forceinline__ int kswap(int s) {
  return (s & ~24) | ((s & 8) << 1) | ((s & 16) >> 1);
}

union frag_u { uint4 q[2]; v16bf v; };

static __device__ __forceinline__ v16bf load_frag(const unsigned short* p) {
  frag_u u;
  u.q[0] = ((const uint4*)p)[0];
  u.q[1] = ((const uint4*)p)[1];
  return u.v;
}

// ---------- gated bilinear resize (align_corners=False), fp32 -> fp32 ----------
// all dims are powers of two: li=log2(ih=iw), lo=log2(oh=ow); scale = ih/oh exact
__global__ void k_resize_f32(const float* __restrict__ src, float* __restrict__ dst,
                             const float* __restrict__ gate,
                             int planes, int li, int lo, float scale) {
  int id = blockIdx.x * blockDim.x + threadIdx.x;
  int total = planes << (2 * lo);
  if (id >= total) return;
  const int iw = 1 << li, ow = 1 << lo;
  int ox = id & (ow - 1); int oy = (id >> lo) & (ow - 1); int p = id >> (2 * lo);
  float sy = (oy + 0.5f) * scale - 0.5f;
  sy = fminf(fmaxf(sy, 0.f), (float)(iw - 1));
  int y0 = (int)floorf(sy); int y1 = min(y0 + 1, iw - 1); float wy = sy - (float)y0;
  float sx = (ox + 0.5f) * scale - 0.5f;
  sx = fminf(fmaxf(sx, 0.f), (float)(iw - 1));
  int x0 = (int)floorf(sx); int x1 = min(x0 + 1, iw - 1); float wx = sx - (float)x0;
  const float* s = src + ((size_t)p << (2 * li));
  float v = s[(y0 << li) + x0] * (1 - wy) * (1 - wx) + s[(y0 << li) + x1] * (1 - wy) * wx
          + s[(y1 << li) + x0] * wy * (1 - wx) + s[(y1 << li) + x1] * wy * wx;
  dst[id] = v * gate[p];
}

// ---------- gated bilinear upsample to 128x128, fp32 -> bf16 ----------
__global__ void k_resize_bf16(const float* __restrict__ src, unsigned short* __restrict__ dst,
                              const float* __restrict__ gate,
                              int planes, int li, float scale) {
  int id = blockIdx.x * blockDim.x + threadIdx.x;
  int total = planes * IH * IW;
  if (id >= total) return;
  const int iw = 1 << li;
  int ox = id & (IW - 1); int oy = (id >> 7) & (IH - 1); int p = id >> 14;
  float sy = (oy + 0.5f) * scale - 0.5f;
  sy = fminf(fmaxf(sy, 0.f), (float)(iw - 1));
  int y0 = (int)floorf(sy); int y1 = min(y0 + 1, iw - 1); float wy = sy - (float)y0;
  float sx = (ox + 0.5f) * scale - 0.5f;
  sx = fminf(fmaxf(sx, 0.f), (float)(iw - 1));
  int x0 = (int)floorf(sx); int x1 = min(x0 + 1, iw - 1); float wx = sx - (float)x0;
  const float* s = src + ((size_t)p << (2 * li));
  float v = s[(y0 << li) + x0] * (1 - wy) * (1 - wx) + s[(y0 << li) + x1] * (1 - wy) * wx
          + s[(y1 << li) + x0] * wy * (1 - wx) + s[(y1 << li) + x1] * wy * wx;
  dst[id] = f2bf_bits(v * gate[p]);
}

// ---------- weight prep: offset-conv weights, packed per-lane fragments ----------
// layout: [L][nt=2][kt=18][lane=32][e=16] bf16, N zero-padded 18 -> 32
__global__ void k_prep_offw(const float* __restrict__ ow, unsigned short* __restrict__ dst) {
  int id = blockIdx.x * blockDim.x + threadIdx.x;
  if (id >= LVL * 2 * NKT_D * 32 * 16) return;
  int e = id & 15; int lane = (id >> 4) & 31; int kt = (id >> 9) % NKT_D;
  int nt = (id / (512 * NKT_D)) & 1; int l = id / (512 * NKT_D * 2);
  int n = nt * 16 + (lane & 15);
  int K = kt * 32 + kswap((lane >> 4) * 16 + e); // table order, matches K-swapped A
  int c = K / 9, k = K - c * 9;
  float v = (n < 18) ? ow[(((size_t)l * 18 + n) * CCH + c) * 9 + k] : 0.f;
  dst[id] = f2bf_bits(v);
}

// ---------- weight prep: deform weights, packed per-lane fragments ----------
// layout: [L][nt=4][kt=18][lane=32][e=16] bf16
__global__ void k_prep_dcw(const float* __restrict__ dcw, unsigned short* __restrict__ dst) {
  int id = blockIdx.x * blockDim.x + threadIdx.x;
  if (id >= LVL * 4 * NKT_D * 32 * 16) return;
  int e = id & 15; int lane = (id >> 4) & 31; int kt = (id >> 9) % NKT_D;
  int nt = (id / (512 * NKT_D)) & 3; int l = id / (512 * NKT_D * 4);
  int n = nt * 16 + (lane & 15);
  int K = kt * 32 + kswap((lane >> 4) * 16 + e);
  int c = K / 9, k = K - c * 9;
  dst[id] = f2bf_bits(dcw[(((size_t)l * OCH + n) * CCH + c) * 9 + k]);
}

// ---------- weight prep: cumulative fuse weights (folds FPN top-down adds) ----------
// layout: [nt=4][kt=8][lane=32][e=16] bf16
__global__ void k_prep_wacc(const float* __restrict__ fw, unsigned short* __restrict__ dst) {
  int id = blockIdx.x * blockDim.x + threadIdx.x;
  if (id >= 4 * NKT_F * 32 * 16) return;
  int e = id & 15; int lane = (id >> 4) & 31; int kt = (id >> 9) % NKT_F;
  int nt = id / (512 * NKT_F);
  int n = nt * 16 + (lane & 15);
  int K = kt * 32 + kswap((lane >> 4) * 16 + e);
  int j = K >> 6, c = K & 63;
  float s = 0.f;
  for (int i = 0; i <= j; ++i) s += fw[(size_t)n * KFUSE + i * 64 + c];
  dst[id] = f2bf_bits(s);
}

// ---------- fused level kernel: offset conv WMMA -> deform conv WMMA ----------
// 256 threads = 8 waves, one 32-pixel tile. Offsets never leave LDS.
__global__ void __launch_bounds__(256)
k_level_wmma(const float* __restrict__ cur,
             const unsigned short* __restrict__ owbp, const float* __restrict__ obias,
             const unsigned short* __restrict__ dwbp, const float* __restrict__ dbias,
             float* __restrict__ out, int lw) {
  __shared__ unsigned short sA[32][KDIM + 8];   // im2col tile, reused by both GEMMs
  __shared__ float s_off[32][20];               // 18 offset channels per pixel (+pad)
  __shared__ int   s_y0[NTAPS], s_x0[NTAPS];
  __shared__ float s_wy[NTAPS], s_wx[NTAPS];

  const int tid  = threadIdx.x;
  const int w    = 1 << lw, h = w;
  const int hw   = 1 << (2 * lw);
  const int pb   = blockIdx.x << 5;
  const int b    = pb >> (2 * lw);
  const int pix0 = pb & (hw - 1);

  // Phase 1: plain im2col A tile (zero-padded 3x3 taps), K-swapped slots
  for (int e = tid; e < 32 * KDIM; e += 256) {
    int p = e & 31, slot = e >> 5;
    int K = kswap(slot);
    int c = K / 9, k = K - c * 9;
    int pix = pix0 + p;
    int y = pix >> lw, x = pix & (w - 1);
    int iy = y + (k / 3 - 1), ix = x + (k % 3 - 1);
    float v = (iy >= 0 && iy < h && ix >= 0 && ix < w)
                ? cur[(((size_t)b * CCH + c) << (2 * lw)) + (iy << lw) + ix] : 0.f;
    sA[p][slot] = f2bf_bits(v);
  }
  __syncthreads();

  const int wave = tid >> 5, lane = tid & 31;
  const int lm = lane & 15, hi = lane >> 4;

  // Phase 2: offset conv GEMM (N=18 padded to 32). Only 4 distinct tiles exist;
  // all 8 waves compute (EXEC stays all-ones for WMMA), waves 0-3 commit.
  {
    const int mh2 = (wave >> 1) & 1, nt2 = wave & 1;
    const int m = mh2 * 16 + lm, n = nt2 * 16 + lm;
    v8f acc = {};
#pragma unroll 3
    for (int kt = 0; kt < NKT_D; ++kt) {
      v16bf a  = load_frag(&sA[m][kt * 32 + hi * 16]);
      v16bf bb = load_frag(owbp + (((size_t)nt2 * NKT_D + kt) * 32 + lane) * 16);
      acc = __builtin_amdgcn_wmma_f32_16x16x32_bf16(false, a, false, bb,
                                                    (short)0, acc, false, false);
    }
    const float bo = (n < 18) ? obias[n] : 0.f;
    if (wave < 4 && n < 18) {
#pragma unroll
      for (int r = 0; r < 8; ++r) {
        int M = mh2 * 16 + hi * 8 + r;
        s_off[M][n] = acc[r] + bo;
      }
    }
  }
  __syncthreads();

  // Phase 3: interp params for 32 pixels x 9 taps from LDS offsets
  for (int t = tid; t < NTAPS; t += 256) {
    int p = t / 9, k = t - p * 9;
    int pix = pix0 + p;
    int y = pix >> lw, x = pix & (w - 1);
    float py = (float)y + (float)(k / 3 - 1) + s_off[p][2 * k];
    float px = (float)x + (float)(k % 3 - 1) + s_off[p][2 * k + 1];
    float fy = floorf(py), fx = floorf(px);
    s_y0[t] = (int)fy; s_x0[t] = (int)fx;
    s_wy[t] = py - fy; s_wx[t] = px - fx;
  }
  __syncthreads();

  // Phase 4: bilinear-gather deform A tile (overwrite sA)
  for (int e = tid; e < 32 * KDIM; e += 256) {
    int p = e & 31, slot = e >> 5;
    int K = kswap(slot);
    int c = K / 9, k = K - c * 9;
    int t = p * 9 + k;
    int y0 = s_y0[t], x0 = s_x0[t];
    float wy = s_wy[t], wx = s_wx[t];
    const float* pl = cur + (((size_t)b * CCH + c) << (2 * lw));
    int y1 = y0 + 1, x1 = x0 + 1;
    int cy0 = min(max(y0, 0), h - 1), cy1 = min(max(y1, 0), h - 1);
    int cx0 = min(max(x0, 0), w - 1), cx1 = min(max(x1, 0), w - 1);
    bool yi0 = (y0 >= 0) && (y0 < h), yi1 = (y1 >= 0) && (y1 < h);
    bool xi0 = (x0 >= 0) && (x0 < w), xi1 = (x1 >= 0) && (x1 < w);
    float v00 = (yi0 && xi0) ? pl[(cy0 << lw) + cx0] : 0.f;
    float v01 = (yi0 && xi1) ? pl[(cy0 << lw) + cx1] : 0.f;
    float v10 = (yi1 && xi0) ? pl[(cy1 << lw) + cx0] : 0.f;
    float v11 = (yi1 && xi1) ? pl[(cy1 << lw) + cx1] : 0.f;
    float val = v00 * (1 - wy) * (1 - wx) + v01 * (1 - wy) * wx
              + v10 * wy * (1 - wx) + v11 * wy * wx;
    sA[p][slot] = f2bf_bits(val);
  }
  __syncthreads();

  // Phase 5: deform GEMM, 18 x v_wmma_f32_16x16x32_bf16
  {
    const int mh = wave >> 2, nt = wave & 3;
    const int m = mh * 16 + lm, n = nt * 16 + lm;
    v8f acc = {};
#pragma unroll 3
    for (int kt = 0; kt < NKT_D; ++kt) {
      v16bf a  = load_frag(&sA[m][kt * 32 + hi * 16]);
      v16bf bb = load_frag(dwbp + (((size_t)nt * NKT_D + kt) * 32 + lane) * 16);
      acc = __builtin_amdgcn_wmma_f32_16x16x32_bf16(false, a, false, bb,
                                                    (short)0, acc, false, false);
    }
    const float bo = dbias[n];
#pragma unroll
    for (int r = 0; r < 8; ++r) {
      int M = mh * 16 + hi * 8 + r;
      out[(((size_t)b * OCH + n) << (2 * lw)) + pix0 + M] = acc[r] + bo;
    }
  }
}

// ---------- SE: mean over H*W per (b,c) ----------
__global__ void k_mean(const float* __restrict__ src, float* __restrict__ mean,
                       int hw, float inv_hw) {
  __shared__ float sb[256];
  const float* p = src + (size_t)blockIdx.x * hw;
  float s = 0.f;
  for (int i = threadIdx.x; i < hw; i += 256) s += p[i];
  sb[threadIdx.x] = s; __syncthreads();
  for (int o = 128; o > 0; o >>= 1) {
    if (threadIdx.x < o) sb[threadIdx.x] += sb[threadIdx.x + o];
    __syncthreads();
  }
  if (threadIdx.x == 0) mean[blockIdx.x] = sb[0] * inv_hw;
}

// ---------- SE: tiny MLP gate ----------
__global__ void k_gate(const float* __restrict__ mean, const float* __restrict__ w1,
                       const float* __restrict__ b1, const float* __restrict__ w2,
                       const float* __restrict__ b2, float* __restrict__ gate) {
  __shared__ float sm[64], sh[4];
  int b = blockIdx.x, c = threadIdx.x;
  sm[c] = mean[b * 64 + c];
  __syncthreads();
  if (c < 4) {
    float s = b1[c];
    for (int j = 0; j < 64; ++j) s += w1[c * 64 + j] * sm[j];
    sh[c] = fmaxf(s, 0.f);
  }
  __syncthreads();
  float s = b2[c];
#pragma unroll
  for (int r = 0; r < 4; ++r) s += w2[c * 4 + r] * sh[r];
  gate[b * 64 + c] = 1.f / (1.f + expf(-s));
}

// ---------- fused 1x1 conv (K=256 GEMM) on WMMA bf16 ----------
__global__ void __launch_bounds__(256)
k_fuse_wmma(const unsigned short* __restrict__ up, const unsigned short* __restrict__ wacc,
            const float* __restrict__ bias, float* __restrict__ out) {
  __shared__ unsigned short sA[32][KFUSE + 8];
  const int tid  = threadIdx.x;
  const int hw   = IH * IW;
  const int pb   = blockIdx.x * 32;
  const int b    = pb / hw;
  const int pix0 = pb - b * hw;

  for (int e = tid; e < 32 * KFUSE; e += 256) {
    int p = e & 31, slot = e >> 5;
    int K = kswap(slot);
    int j = K >> 6, c = K & 63;
    sA[p][slot] = up[(((size_t)j * BSZ + b) * CCH + c) * hw + pix0 + p];
  }
  __syncthreads();

  const int wave = tid >> 5, lane = tid & 31;
  const int mh = wave >> 2, nt = wave & 3;
  const int lm = lane & 15, hi = lane >> 4;
  const int m  = mh * 16 + lm, n = nt * 16 + lm;
  v8f acc = {};
#pragma unroll
  for (int kt = 0; kt < NKT_F; ++kt) {
    v16bf a  = load_frag(&sA[m][kt * 32 + hi * 16]);
    v16bf bb = load_frag(wacc + (((size_t)nt * NKT_F + kt) * 32 + lane) * 16);
    acc = __builtin_amdgcn_wmma_f32_16x16x32_bf16(false, a, false, bb,
                                                  (short)0, acc, false, false);
  }
  const float bo = bias[n];
#pragma unroll
  for (int r = 0; r < 8; ++r) {
    int M = mh * 16 + hi * 8 + r;
    out[((size_t)b * OCH + n) * hw + pix0 + M] = acc[r] + bo;
  }
}

// ---------- host orchestration ----------
extern "C" void kernel_launch(void* const* d_in, const int* in_sizes, int n_in,
                              void* d_out, int out_size, void* d_ws, size_t ws_size,
                              hipStream_t stream) {
  (void)in_sizes; (void)n_in; (void)out_size; (void)ws_size;
  const float* x      = (const float*)d_in[0];
  const float* off_w  = (const float*)d_in[1];
  const float* off_b  = (const float*)d_in[2];
  const float* dc_w   = (const float*)d_in[3];
  const float* dc_b   = (const float*)d_in[4];
  const float* se_w1  = (const float*)d_in[5];
  const float* se_b1  = (const float*)d_in[6];
  const float* se_w2  = (const float*)d_in[7];
  const float* se_b2  = (const float*)d_in[8];
  const float* fuse_w = (const float*)d_in[9];
  const float* fuse_b = (const float*)d_in[10];
  float* out = (float*)d_out;

  char* base = (char*)d_ws;
  size_t cur_off = 0;
  auto alloc = [&](size_t bytes) -> char* {
    char* p = base + cur_off;
    cur_off = (cur_off + bytes + 255) & ~(size_t)255;
    return p;
  };
  const int lws[LVL] = {7, 6, 5, 4};             // log2 of level width
  float* curb[LVL]; curb[0] = nullptr;
  for (int l = 1; l < LVL; ++l)
    curb[l] = (float*)alloc(sizeof(float) * ((size_t)BSZ * CCH << (2 * lws[l])));
  float* dcb[LVL];
  for (int l = 0; l < LVL; ++l)
    dcb[l] = (float*)alloc(sizeof(float) * ((size_t)BSZ * OCH << (2 * lws[l])));
  float* mean = (float*)alloc(sizeof(float) * BSZ * OCH);
  float* gate = (float*)alloc(sizeof(float) * BSZ * OCH);
  unsigned short* offwb = (unsigned short*)alloc(sizeof(unsigned short) * LVL * 2 * KDIM * 32);
  unsigned short* dcwb  = (unsigned short*)alloc(sizeof(unsigned short) * LVL * KDIM * OCH);
  unsigned short* wacc  = (unsigned short*)alloc(sizeof(unsigned short) * KFUSE * OCH);
  unsigned short* up    = (unsigned short*)alloc(sizeof(unsigned short) * (size_t)LVL * BSZ * CCH * IH * IW);

  k_prep_offw<<<(LVL * 2 * NKT_D * 512 + 255) / 256, 256, 0, stream>>>(off_w, offwb);
  k_prep_dcw<<<(LVL * 4 * NKT_D * 512 + 255) / 256, 256, 0, stream>>>(dc_w, dcwb);
  k_prep_wacc<<<(4 * NKT_F * 512 + 255) / 256, 256, 0, stream>>>(fuse_w, wacc);

  for (int l = 0; l < LVL; ++l) {
    const int lw = lws[l], hw = 1 << (2 * lw);
    const float* cur;
    if (l == 0) {
      cur = x;
    } else {
      int tot = (BSZ * CCH) << (2 * lw);
      // gate still holds level (l-1)'s gate here -> scaled downsample
      k_resize_f32<<<(tot + 255) / 256, 256, 0, stream>>>(dcb[l - 1], curb[l], gate,
                                                          BSZ * CCH, lws[l - 1], lw, 2.0f);
      cur = curb[l];
    }
    k_level_wmma<<<(BSZ * hw) / 32, 256, 0, stream>>>(
        cur, offwb + (size_t)l * 2 * NKT_D * 512, off_b + l * 18,
        dcwb + (size_t)l * 4 * NKT_D * 512, dc_b + l * OCH, dcb[l], lw);
    k_mean<<<BSZ * OCH, 256, 0, stream>>>(dcb[l], mean, hw, 1.0f / (float)hw);
    k_gate<<<BSZ, 64, 0, stream>>>(mean, se_w1 + l * 4 * OCH, se_b1 + l * 4,
                                   se_w2 + l * OCH * 4, se_b2 + l * OCH, gate);
    int tot_up = BSZ * CCH * IH * IW;
    k_resize_bf16<<<(tot_up + 255) / 256, 256, 0, stream>>>(
        dcb[l], up + (size_t)l * BSZ * CCH * IH * IW, gate, BSZ * CCH,
        lw, (float)(1 << lw) / 128.0f);
  }
  k_fuse_wmma<<<(BSZ * IH * IW) / 32, 256, 0, stream>>>(up, wacc, fuse_b, out);
}
